// Attention_88940182766421
// MI455X (gfx1250) — matrix operand order repaired and checked by reading, not compiled, not run on hardware
//
#include <hip/hip_runtime.h>
#include <cmath>

// ---------- problem constants ----------
constexpr int BB = 2, TT = 2048, DD = 3584, NQ = 16, NKV = 8, HH = 256;
constexpr int M_ROWS   = BB * TT;                 // 4096
constexpr int QKV_COLS = NQ * HH + 2 * NKV * HH;  // 8192
constexpr int ENC_K    = NQ * HH;                 // 4096 (reduction dim of o-proj)
constexpr float SOFT_CAP = 50.0f;
constexpr float SCALAR   = 0.0625f;
constexpr int   WIN      = 1024;

// ---------- vector types (trivial, union-safe) ----------
typedef __attribute__((ext_vector_type(16))) __bf16   v16bf;
typedef __attribute__((ext_vector_type(8)))  float    v8f;
typedef __attribute__((ext_vector_type(4)))  unsigned u32x4;

#define WMMA_BF16(a, b, c) \
  __builtin_amdgcn_wmma_f32_16x16x32_bf16(false, (a), false, (b), (short)0, (c), false, false)

// ---------- fragment loaders (CDNA5 16-bit WMMA layouts, ISA 7.12.2) ----------
// A-matrix 16x32 bf16: lane m = lane&15, g = lane>>4.
// VGPR p<4 : K = 2p + 8g ; VGPR p>=4 : K = 16 + 2(p-4) + 8g  (two bf16 per VGPR)
__device__ inline v16bf load_a_frag(const __bf16* row32, int g) {
  union { v16bf v; unsigned u[8]; } r;
  const unsigned short* hp = (const unsigned short*)row32;
#pragma unroll
  for (int p = 0; p < 8; ++p) {
    int k = ((p < 4) ? (2 * p) : (16 + 2 * (p - 4))) + 8 * g;
    r.u[p] = *(const unsigned*)(hp + k);  // even k -> 4B aligned pair load
  }
  return r.v;
}

// B-matrix 32x16 bf16: lane holds column N=lane&15, element s -> K = s + 8g (s<8), 16 + (s-8) + 8g (s>=8)
__device__ inline v16bf load_b_frag(const __bf16* p, int ld, int g) {
  union { v16bf v; __bf16 e[16]; } r;
#pragma unroll
  for (int s = 0; s < 16; ++s) {
    int k = ((s < 8) ? s : (8 + s)) + 8 * g;
    r.e[s] = p[k * ld];
  }
  return r.v;
}

// =====================================================================
// Kernel 1/5: tiled bf16 WMMA GEMM.
// MODE 0: qkv = x(4096x3584,f32) @ [q_w | kv_w](3584x8192)      -> f32
// MODE 1: out = enc(4096x4096,bf16) @ o_w(4096x3584,f32->bf16)  -> f32
// Block: 256 threads (8 waves). Tile: 128 rows x 64 cols, k-step 32.
// =====================================================================
template <int MODE>
__global__ __launch_bounds__(256, 1)
void gemm_kernel(const float* __restrict__ xf,
                 const unsigned short* __restrict__ abf_us,
                 const float* __restrict__ qw,
                 const float* __restrict__ kvw,
                 const float* __restrict__ ow,
                 float* __restrict__ out) {
  constexpr int KRED = (MODE == 0) ? DD : ENC_K;       // 3584 / 4096
  constexpr int NC   = (MODE == 0) ? QKV_COLS : DD;    // 8192 / 3584

  const int col0 = blockIdx.x * 64;
  const int row0 = blockIdx.y * 128;
  const int tid  = threadIdx.x;
  const int wv   = tid >> 5;
  const int ln   = tid & 31;
  const int g    = ln >> 4;
  const int ml   = ln & 15;

  __shared__ __bf16 As[128][32];
  __shared__ __bf16 Ws[32][64];

  const float* wbase = nullptr;
  int h0 = 0;
  if (MODE == 0) {
    const int head = col0 >> 8;  // 64-col tile never crosses a 256-col head
    wbase = (col0 < NQ * HH) ? (qw + (size_t)head * DD * HH)
                             : (kvw + (size_t)(head - NQ) * DD * HH);
    h0 = col0 & 255;
  }
  const __bf16* abf = (const __bf16*)abf_us;

  v8f acc[4];
#pragma unroll
  for (int ct = 0; ct < 4; ++ct) acc[ct] = v8f{0, 0, 0, 0, 0, 0, 0, 0};

  for (int d0 = 0; d0 < KRED; d0 += 32) {
    __syncthreads();
    // A tile 128x32 -> bf16
#pragma unroll
    for (int i = 0; i < 16; ++i) {
      int idx = tid + i * 256;
      int r = idx >> 5, dd = idx & 31;
      if (MODE == 0)
        As[r][dd] = (__bf16)xf[(size_t)(row0 + r) * DD + d0 + dd];
      else
        As[r][dd] = abf[(size_t)(row0 + r) * ENC_K + d0 + dd];
    }
    // W tile 32x64 -> bf16
#pragma unroll
    for (int i = 0; i < 8; ++i) {
      int idx = tid + i * 256;
      int dd = idx >> 6, cc = idx & 63;
      float w = (MODE == 0) ? wbase[(size_t)(d0 + dd) * HH + h0 + cc]
                            : ow[(size_t)(d0 + dd) * DD + col0 + cc];
      Ws[dd][cc] = (__bf16)w;
    }
    __syncthreads();

    v16bf a = load_a_frag(&As[wv * 16 + ml][0], g);
#pragma unroll
    for (int ct = 0; ct < 4; ++ct) {
      v16bf b = load_b_frag(&Ws[0][ct * 16 + ml], 64, g);
      acc[ct] = WMMA_BF16(a, b, acc[ct]);
    }
  }

  // D layout: lane holds col = ml, rows m = r + 8g
#pragma unroll
  for (int ct = 0; ct < 4; ++ct)
#pragma unroll
    for (int r = 0; r < 8; ++r) {
      int m = wv * 16 + r + 8 * g;
      int c = col0 + ct * 16 + ml;
      out[(size_t)(row0 + m) * NC + c] = acc[ct][r];
    }
}

// =====================================================================
// Kernel 2: RoPE (+ q scale) -> bf16 q,k. One thread per (b,t,head,hpair)
// heads 0..15 = q, 16..23 = k.  qkv row layout: [q(4096) | k(2048) | v(2048)]
// =====================================================================
__global__ void rope_kernel(const float* __restrict__ qkv,
                            const int* __restrict__ segpos,
                            unsigned short* __restrict__ qb_us,
                            unsigned short* __restrict__ kb_us) {
  __bf16* qb = (__bf16*)qb_us;
  __bf16* kb = (__bf16*)kb_us;
  int idx = blockIdx.x * 256 + threadIdx.x;  // < B*T*24*128
  int hp   = idx & 127;
  int head = (idx >> 7) % 24;
  int bt   = idx / 3072;
  if (bt >= BB * TT) return;

  float pos = (float)segpos[bt];
  float ts  = powf(10000.0f, (float)hp * (1.0f / 128.0f));
  float ang = pos / ts;
  float sn = sinf(ang), cs = cosf(ang);
  const float* row = qkv + (size_t)bt * QKV_COLS;

  if (head < NQ) {
    int c1 = head * HH + hp;
    float x1 = row[c1], x2 = row[c1 + 128];
    size_t o = ((size_t)bt * NQ + head) * HH + hp;
    qb[o]       = (__bf16)((x1 * cs - x2 * sn) * SCALAR);
    qb[o + 128] = (__bf16)((x2 * cs + x1 * sn) * SCALAR);
  } else {
    int kk = head - NQ;
    int c1 = NQ * HH + kk * HH + hp;
    float x1 = row[c1], x2 = row[c1 + 128];
    size_t o = ((size_t)bt * NKV + kk) * HH + hp;
    kb[o]       = (__bf16)(x1 * cs - x2 * sn);
    kb[o + 128] = (__bf16)(x2 * cs + x1 * sn);
  }
}

// Kernel 3: v -> bf16
__global__ void vconv_kernel(const float* __restrict__ qkv,
                             unsigned short* __restrict__ vb_us) {
  __bf16* vb = (__bf16*)vb_us;
  int idx = blockIdx.x * 256 + threadIdx.x;  // < B*T*8*256
  if (idx >= BB * TT * NKV * HH) return;
  int h  = idx & 255;
  int kk = (idx >> 8) & 7;
  int bt = idx >> 11;
  vb[idx] = (__bf16)qkv[(size_t)bt * QKV_COLS + (NQ * HH + NKV * HH) + kk * HH + h];
}

// =====================================================================
// Kernel 4: sliding-window flash attention.
// Block = 64 threads (2 waves) = one kv head's two grouped q heads.
// K/V double-buffered in LDS, filled with CDNA5 async global->LDS copies
// (GLOBAL_LOAD_ASYNC_TO_LDS_B128, tracked by ASYNCcnt): chunk i+1 streams
// in while WMMAs consume chunk i. One barrier + one s_wait_asynccnt/chunk.
// =====================================================================
__global__ __launch_bounds__(64, 1)
void attn_kernel(const unsigned short* __restrict__ qb_us,
                 const unsigned short* __restrict__ kb_us,
                 const unsigned short* __restrict__ vb_us,
                 unsigned short* __restrict__ encb_us) {
  const __bf16* qb = (const __bf16*)qb_us;
  const __bf16* kb = (const __bf16*)kb_us;
  const __bf16* vb = (const __bf16*)vb_us;
  __bf16* encb = (__bf16*)encb_us;

  const int t0  = blockIdx.x * 16;
  const int kk  = blockIdx.y;
  const int b   = blockIdx.z;
  const int tid = threadIdx.x;
  const int wv  = tid >> 5;       // 0/1 -> q head n = 2*kk + wv
  const int n   = 2 * kk + wv;
  const int ln  = tid & 31;
  const int g   = ln >> 4;
  const int ml  = ln & 15;

  __shared__ __bf16 Ks[2][32][HH];     // 32 KB, double-buffered
  __shared__ __bf16 Vs[2][32][HH];     // 32 KB, double-buffered
  __shared__ __bf16 Qt[2][HH][16];     // 16 KB: Q^T per wave [h][query]
  __shared__ float  Ls[2][16][32];     // logits [query][key], per wave
  __shared__ __bf16 Ps[2][16][32];     // probs, per wave
  __shared__ float  alpha_s[2][16];
  __shared__ float  inv_s[2][16];

  // Stage Q^T once: lane handles query row ml, h-range [g*128, g*128+128)
  {
    const __bf16* qsrc = qb + ((size_t)(b * TT + t0 + ml) * NQ + n) * HH + g * 128;
#pragma unroll 8
    for (int i = 0; i < 128; ++i) Qt[wv][g * 128 + i][ml] = qsrc[i];
  }

  v8f oacc[16];
#pragma unroll
  for (int t = 0; t < 16; ++t) oacc[t] = v8f{0, 0, 0, 0, 0, 0, 0, 0};

  float m_i = -3.4e38f, l_i = 0.0f;  // valid in lanes 0..15 (row = ln)
  const int tq = t0 + ln;

  int s_lo = t0 - (WIN - 1);
  if (s_lo < 0) s_lo = 0;
  const int s_begin = s_lo & ~31;
  const int s_end   = t0 + 16;
  const int nchunks = (s_end - s_begin + 31) >> 5;

  // Async-stage one 32-key K/V chunk into LDS buffer `buf`.
  // Wave wv covers keys wv*16 .. wv*16+15; each instruction moves one
  // full 512B key row (lane = 16B slice). Keys past s_end are clamped
  // (finite data; their P columns are exactly 0 after masking).
  auto stage = [&](int s0c, int buf) {
#pragma unroll
    for (int i = 0; i < 16; ++i) {
      int key = wv * 16 + i;
      int s = s0c + key;
      if (s > s_end - 1) s = s_end - 1;
      const __bf16* gk = kb + ((size_t)(b * TT + s) * NKV + kk) * HH + ln * 8;
      const __bf16* gv = vb + ((size_t)(b * TT + s) * NKV + kk) * HH + ln * 8;
      unsigned dk = (unsigned)(size_t)&Ks[buf][key][ln * 8];  // LDS byte offset
      unsigned dv = (unsigned)(size_t)&Vs[buf][key][ln * 8];
      asm volatile("global_load_async_to_lds_b128 %0, %1, off"
                   :: "v"(dk), "v"(gk) : "memory");
      asm volatile("global_load_async_to_lds_b128 %0, %1, off"
                   :: "v"(dv), "v"(gv) : "memory");
    }
  };

  stage(s_begin, 0);  // prologue: chunk 0 in flight

  for (int ci = 0; ci < nchunks; ++ci) {
    const int s0  = s_begin + ci * 32;
    const int buf = ci & 1;

    // my asyncs (filling `buf`) have landed in LDS...
    asm volatile("s_wait_asynccnt 0x0" ::: "memory");
    // ...and after the barrier, both waves' have; previous compute is done,
    // so buf^1 is free to be overwritten by the prefetch.
    __syncthreads();
    if (ci + 1 < nchunks) stage(s0 + 32, buf ^ 1);

    // logits^T tile: D = K(16 keys x 32h) x Q^T(32h x 16 queries)
    // D layout: lane col = query ml, element r -> key-local r + 8g.
#pragma unroll
    for (int ct = 0; ct < 2; ++ct) {
      v8f lacc = v8f{0, 0, 0, 0, 0, 0, 0, 0};
#pragma unroll
      for (int hc = 0; hc < 8; ++hc) {
        v16bf ak = load_a_frag(&Ks[buf][ct * 16 + ml][hc * 32], g);
        v16bf bq = load_b_frag(&Qt[wv][hc * 32][ml], 16, g);
        lacc = WMMA_BF16(ak, bq, lacc);
      }
#pragma unroll
      for (int r = 0; r < 8; ++r) Ls[wv][ml][ct * 16 + r + 8 * g] = lacc[r];
    }

    // online softmax with soft-cap + sliding-window causal mask (row = ln)
    if (ln < 16) {
      float cmax = -3.4e38f;
#pragma unroll
      for (int j = 0; j < 32; ++j) {
        int s = s0 + j;
        float v = Ls[wv][ln][j];
        v = tanhf(v * (1.0f / SOFT_CAP)) * SOFT_CAP;
        bool ok = (s <= tq) && (s >= tq - (WIN - 1));
        v = ok ? v : -3.4e38f;
        Ls[wv][ln][j] = v;               // write back capped+masked value
        cmax = (v > cmax) ? v : cmax;
      }
      float m_new = m_i > cmax ? m_i : cmax;
      float al = expf(m_i - m_new);  // ==1 when equal; underflows to 0 otherwise
      float sum = 0.0f;
#pragma unroll
      for (int j = 0; j < 32; ++j) {
        float lgv = Ls[wv][ln][j];
        float p = (lgv > -1e37f) ? expf(lgv - m_new) : 0.0f;  // all-masked-safe
        Ps[wv][ln][j] = (__bf16)p;
        sum += p;
      }
      l_i = l_i * al + sum;
      m_i = m_new;
      alpha_s[wv][ln] = al;
    }

    // rescale O accumulators by alpha (rows m = r + 8g)
    float ar[8];
#pragma unroll
    for (int r = 0; r < 8; ++r) ar[r] = alpha_s[wv][r + 8 * g];
#pragma unroll
    for (int t = 0; t < 16; ++t)
#pragma unroll
      for (int r = 0; r < 8; ++r) oacc[t][r] *= ar[r];

    // O += P . V  (A = P 16x32, B = V rows from LDS: lane = h column t*16+ml, elements = keys)
    v16bf pf = load_a_frag(&Ps[wv][ml][0], g);
#pragma unroll
    for (int t = 0; t < 16; ++t) {
      v16bf bv = load_b_frag(&Vs[buf][0][t * 16 + ml], HH, g);
      oacc[t] = WMMA_BF16(pf, bv, oacc[t]);
    }
  }

  if (ln < 16) inv_s[wv][ln] = 1.0f / l_i;  // diagonal always attended -> l_i > 0

#pragma unroll
  for (int r = 0; r < 8; ++r) {
    float inv = inv_s[wv][r + 8 * g];
    int t = t0 + r + 8 * g;
    __bf16* orow = encb + ((size_t)(b * TT + t) * NQ + n) * HH;
#pragma unroll
    for (int tile = 0; tile < 16; ++tile)
      orow[tile * 16 + ml] = (__bf16)(oacc[tile][r] * inv);
  }
}

// =====================================================================
// Host launcher
// =====================================================================
extern "C" void kernel_launch(void* const* d_in, const int* in_sizes, int n_in,
                              void* d_out, int out_size, void* d_ws, size_t ws_size,
                              hipStream_t stream) {
  const float* x     = (const float*)d_in[0];
  const int* segpos  = (const int*)d_in[1];
  // d_in[2] = attn_mask (bool) — derived analytically, unused
  const float* q_w   = (const float*)d_in[3];
  const float* kv_w  = (const float*)d_in[4];
  const float* o_w   = (const float*)d_in[5];
  float* out         = (float*)d_out;

  // workspace layout (bytes):
  //   qkv f32 : 4096*8192*4   = 134217728
  //   q  bf16 : 4096*16*256*2 =  33554432
  //   k  bf16 : 4096*8*256*2  =  16777216
  //   v  bf16 : 4096*8*256*2  =  16777216
  //   enc bf16: 4096*16*256*2 =  33554432   (total ~235 MB)
  char* ws = (char*)d_ws;
  float*          qkv  = (float*)ws;
  unsigned short* qb   = (unsigned short*)(ws + 134217728ULL);
  unsigned short* kb   = (unsigned short*)(ws + 134217728ULL + 33554432ULL);
  unsigned short* vb   = (unsigned short*)(ws + 134217728ULL + 50331648ULL);
  unsigned short* encb = (unsigned short*)(ws + 134217728ULL + 67108864ULL);

  // 1) fused QKV projection (WMMA bf16)
  gemm_kernel<0><<<dim3(QKV_COLS / 64, M_ROWS / 128), 256, 0, stream>>>(
      x, nullptr, q_w, kv_w, nullptr, qkv);

  // 2) RoPE + scale -> bf16 q,k ; 3) v -> bf16
  rope_kernel<<<(BB * TT * 24 * 128) / 256, 256, 0, stream>>>(qkv, segpos, qb, kb);
  vconv_kernel<<<(BB * TT * NKV * HH) / 256, 256, 0, stream>>>(qkv, vb);

  // 4) sliding-window flash attention (WMMA bf16 + async K/V pipeline)
  attn_kernel<<<dim3(TT / 16, NKV, BB), 64, 0, stream>>>(qb, kb, vb, encb);

  // 5) output projection (WMMA bf16) -> f32 out
  gemm_kernel<1><<<dim3(DD / 64, M_ROWS / 128), 256, 0, stream>>>(
      nullptr, encb, nullptr, nullptr, o_w, out);
}
